// SkipGNN_33492154974252
// MI455X (gfx1250) — compile-verified, matrix operand order checked by reading, not run on hardware
//
#include <hip/hip_runtime.h>

#define N_NODES  50000
#define N_EDGES  800000
#define N_GRAPHS 64

typedef __attribute__((ext_vector_type(16))) __bf16 v16bf;
typedef __attribute__((ext_vector_type(8)))  __bf16 v8bf;
typedef __attribute__((ext_vector_type(8)))  float  v8f;

// ---------------------------------------------------------------------------
// fp32 -> bf16 (round to nearest even)
// ---------------------------------------------------------------------------
__device__ __forceinline__ __bf16 f2bf(float f) {
  unsigned u = __builtin_bit_cast(unsigned, f);
  unsigned short h = (unsigned short)((u + 0x7FFFu + ((u >> 16) & 1u)) >> 16);
  return __builtin_bit_cast(__bf16, h);
}

// Load one WMMA 16x16x32 bf16 fragment (A or B role) from a row-major bf16
// matrix with leading dimension ld.  Lane layout per CDNA5 ISA:
//   lanes 0-15  : row = r, K = {k0+0..7,  k0+16..23}
//   lanes 16-31 : row = r, K = {k0+8..15, k0+24..31}
// Each 8-element run is 16 contiguous bytes -> one b128 load.
__device__ __forceinline__ v16bf load_frag(const __bf16* __restrict__ base,
                                           int ld, int row, int k0, int half) {
  const __bf16* p = base + (size_t)row * ld + k0 + (half << 3);
  v8bf lo = *reinterpret_cast<const v8bf*>(p);
  v8bf hi = *reinterpret_cast<const v8bf*>(p + 16);
  return __builtin_shufflevector(lo, hi, 0, 1, 2, 3, 4, 5, 6, 7,
                                 8, 9, 10, 11, 12, 13, 14, 15);
}

#define WMMA_BF16(A, B, C)                                                    \
  __builtin_amdgcn_wmma_f32_16x16x32_bf16(false, (A), false, (B), (short)0,   \
                                          (C), false, false)

__device__ __forceinline__ void store_tile(v8f acc, const float* __restrict__ bias,
                                           int tm, int tn,
                                           float* __restrict__ Cf,
                                           __bf16* __restrict__ Cb,
                                           int N, int act, int half, int r) {
  float bv = bias ? bias[tn * 16 + r] : 0.0f;
  size_t off = (size_t)(tm * 16 + 8 * half) * N + tn * 16 + r;
#pragma unroll
  for (int v = 0; v < 8; ++v) {
    float val = acc[v] + bv;
    if (act == 1) val = (val >= 0.0f) ? val : 0.01f * val;
    if (Cf) Cf[off + (size_t)v * N] = val;
    if (Cb) Cb[off + (size_t)v * N] = f2bf(val);
  }
}

// ---------------------------------------------------------------------------
// WMMA GEMM: C[M x N] = act( A1[M x K1] @ W1[N x K1]^T
//                          (+ A2[M x K2] @ W2[N x K2]^T) + bias )
// All operands bf16 row-major; fp32 accumulation. One wave computes a
// 32(M) x 32(N) block as 2x2 WMMA tiles (A/B fragments reused twice each).
// Outputs: optional fp32 copy Cf and/or bf16 copy Cb.
// ---------------------------------------------------------------------------
__global__ __launch_bounds__(256)
void gemm_wmma_kernel(const __bf16* __restrict__ A1, const __bf16* __restrict__ W1, int K1,
                      const __bf16* __restrict__ A2, const __bf16* __restrict__ W2, int K2,
                      const float* __restrict__ bias,
                      float* __restrict__ Cf, __bf16* __restrict__ Cb,
                      int M, int N, int act) {
  int tilesM = M >> 4;
  int mG = (tilesM + 1) >> 1;      // 2 M-tiles per wave (last group may be short)
  int nG = N >> 5;                 // 2 N-tiles per wave (N is a multiple of 32)
  int wave = blockIdx.x * (blockDim.x >> 5) + (threadIdx.x >> 5);
  if (wave >= mG * nG) return;     // wave-uniform guard: EXEC stays all-ones
  int mg = wave / nG;
  int ng = wave - mg * nG;

  int lane = threadIdx.x & 31;
  int half = lane >> 4;
  int r    = lane & 15;

  int tm0 = mg * 2;
  int tm1 = tm0 + 1;
  int tm1c = (tm1 < tilesM) ? tm1 : tm0;   // clamp loads for odd tail; store guarded
  int rowA0 = tm0 * 16 + r;
  int rowA1 = tm1c * 16 + r;
  int colB0 = ng * 32 + r;
  int colB1 = colB0 + 16;

  v8f acc00 = {}, acc01 = {}, acc10 = {}, acc11 = {};

  for (int k0 = 0; k0 < K1; k0 += 32) {
    v16bf a0 = load_frag(A1, K1, rowA0, k0, half);
    v16bf a1 = load_frag(A1, K1, rowA1, k0, half);
    v16bf b0 = load_frag(W1, K1, colB0, k0, half);
    v16bf b1 = load_frag(W1, K1, colB1, k0, half);
    acc00 = WMMA_BF16(a0, b0, acc00);
    acc01 = WMMA_BF16(a0, b1, acc01);
    acc10 = WMMA_BF16(a1, b0, acc10);
    acc11 = WMMA_BF16(a1, b1, acc11);
  }
  if (A2 != nullptr) {
    for (int k0 = 0; k0 < K2; k0 += 32) {
      v16bf a0 = load_frag(A2, K2, rowA0, k0, half);
      v16bf a1 = load_frag(A2, K2, rowA1, k0, half);
      v16bf b0 = load_frag(W2, K2, colB0, k0, half);
      v16bf b1 = load_frag(W2, K2, colB1, k0, half);
      acc00 = WMMA_BF16(a0, b0, acc00);
      acc01 = WMMA_BF16(a0, b1, acc01);
      acc10 = WMMA_BF16(a1, b0, acc10);
      acc11 = WMMA_BF16(a1, b1, acc11);
    }
  }

  int tn0 = ng * 2, tn1 = tn0 + 1;
  store_tile(acc00, bias, tm0, tn0, Cf, Cb, N, act, half, r);
  store_tile(acc01, bias, tm0, tn1, Cf, Cb, N, act, half, r);
  if (tm1 < tilesM) {
    store_tile(acc10, bias, tm1, tn0, Cf, Cb, N, act, half, r);
    store_tile(acc11, bias, tm1, tn1, Cf, Cb, N, act, half, r);
  }
}

// ---------------------------------------------------------------------------
// Utility kernels
// ---------------------------------------------------------------------------
__global__ void zero_kernel(float4* __restrict__ p, long long n4) {
  long long i = blockIdx.x * (long long)blockDim.x + threadIdx.x;
  if (i < n4) { float4 z; z.x = z.y = z.z = z.w = 0.0f; p[i] = z; }
}

// fp32 -> bf16 bulk convert, 8 elements per thread (n must be a multiple of 8)
__global__ void cvt_bf16_kernel(const float* __restrict__ in,
                                __bf16* __restrict__ out, long long n8) {
  long long i = blockIdx.x * (long long)blockDim.x + threadIdx.x;
  if (i >= n8) return;
  const float4* p = reinterpret_cast<const float4*>(in) + i * 2;
  float4 a = p[0], b = p[1];
  v8bf r = {f2bf(a.x), f2bf(a.y), f2bf(a.z), f2bf(a.w),
            f2bf(b.x), f2bf(b.y), f2bf(b.z), f2bf(b.w)};
  reinterpret_cast<v8bf*>(out)[i] = r;
}

// agg[dst[e]] += x[src[e]] over all edges, 4 channels per thread
__global__ void scatter_add_kernel(const float* __restrict__ x,
                                   const int* __restrict__ src,
                                   const int* __restrict__ dst,
                                   float* __restrict__ agg, int C4) {
  long long i = blockIdx.x * (long long)blockDim.x + threadIdx.x;
  long long total = (long long)N_EDGES * C4;
  if (i >= total) return;
  int e = (int)(i / C4);
  int c = (int)(i - (long long)e * C4);
  int s = src[e], d = dst[e];
  float4 v = ((const float4*)x)[(size_t)s * C4 + c];
  float* o = agg + ((size_t)d * C4 + c) * 4;
  unsafeAtomicAdd(o + 0, v.x);
  unsafeAtomicAdd(o + 1, v.y);
  unsafeAtomicAdd(o + 2, v.z);
  unsafeAtomicAdd(o + 3, v.w);
}

// concat along channels, writing fp32 and bf16 copies
__global__ void concat_kernel(const float* __restrict__ a, int Ca,
                              const float* __restrict__ b, int Cb,
                              float* __restrict__ outF, __bf16* __restrict__ outB) {
  int C = Ca + Cb;
  long long i = blockIdx.x * (long long)blockDim.x + threadIdx.x;
  long long total = (long long)N_NODES * C;
  if (i >= total) return;
  int node = (int)(i / C);
  int c    = (int)(i - (long long)node * C);
  float v = (c < Ca) ? a[(size_t)node * Ca + c] : b[(size_t)node * Cb + (c - Ca)];
  outF[i] = v;
  outB[i] = f2bf(v);
}

__global__ void pool_scatter_kernel(const float* __restrict__ x,
                                    const int* __restrict__ batch,
                                    float* __restrict__ pool,
                                    float* __restrict__ cnt) {
  long long i = blockIdx.x * (long long)blockDim.x + threadIdx.x;
  if (i >= (long long)N_NODES * 128) return;
  int node = (int)(i >> 7);
  int c    = (int)(i & 127);
  int g    = batch[node];
  unsafeAtomicAdd(&pool[(size_t)g * 128 + c], x[(size_t)node * 128 + c]);
  if (c == 0) unsafeAtomicAdd(&cnt[g], 1.0f);
}

__global__ void pool_div_kernel(float* __restrict__ pool, const float* __restrict__ cnt) {
  int i = blockIdx.x * blockDim.x + threadIdx.x;
  if (i >= N_GRAPHS * 128) return;
  float c = cnt[i >> 7];
  pool[i] /= (c > 1.0f ? c : 1.0f);
}

// tiny dense layer for the MLP head: out[g,o] = relu?(in[g,:]@W[o,:] + b[o])
__global__ void mlp_kernel(const float* __restrict__ in, const float* __restrict__ W,
                           const float* __restrict__ b, float* __restrict__ out,
                           int K, int Nout, int relu) {
  int i = blockIdx.x * blockDim.x + threadIdx.x;
  if (i >= N_GRAPHS * Nout) return;
  int g = i / Nout, o = i - g * Nout;
  const float* ip = in + (size_t)g * K;
  const float* wp = W  + (size_t)o * K;
  float s = b[o];
  for (int k = 0; k < K; ++k) s += ip[k] * wp[k];
  if (relu && s < 0.0f) s = 0.0f;
  out[i] = s;
}

// ---------------------------------------------------------------------------
// Host-side helpers
// ---------------------------------------------------------------------------
static void launch_zero(float* p, size_t nf, hipStream_t s) {
  long long n4 = (long long)(nf / 4);
  zero_kernel<<<(unsigned)((n4 + 255) / 256), 256, 0, s>>>((float4*)p, n4);
}

static void launch_cvt(const float* in, __bf16* out, size_t n, hipStream_t s) {
  long long n8 = (long long)(n / 8);
  cvt_bf16_kernel<<<(unsigned)((n8 + 255) / 256), 256, 0, s>>>(in, out, n8);
}

static void launch_gemm(const __bf16* A1, const __bf16* W1, int K1,
                        const __bf16* A2, const __bf16* W2, int K2,
                        const float* bias, float* Cf, __bf16* Cb,
                        int M, int N, int act, hipStream_t s) {
  int tilesM = M / 16;
  int mG = (tilesM + 1) / 2;
  int nG = N / 32;
  int waves = mG * nG;
  gemm_wmma_kernel<<<(waves + 7) / 8, 256, 0, s>>>(A1, W1, K1, A2, W2, K2,
                                                   bias, Cf, Cb, M, N, act);
}

static void run_layer(const float* xinF, const __bf16* xinB, int Cin, int hc,
                      const __bf16* wrelB, const float* brel, const __bf16* wrootB,
                      const __bf16* wlinB, const float* blin,
                      const int* src, const int* dst,
                      float* agg, __bf16* aggB, __bf16* hB,
                      float* xoutF, __bf16* xoutB, hipStream_t s) {
  launch_zero(agg, (size_t)N_NODES * Cin, s);
  int C4 = Cin / 4;
  long long total = (long long)N_EDGES * C4;
  scatter_add_kernel<<<(unsigned)((total + 255) / 256), 256, 0, s>>>(xinF, src, dst, agg, C4);
  launch_cvt(agg, aggB, (size_t)N_NODES * Cin, s);
  // h = LeakyReLU(agg @ wrel^T + xin @ wroot^T + brel)   (bf16-only output)
  launch_gemm(aggB, wrelB, Cin, xinB, wrootB, Cin, brel, nullptr, hB, N_NODES, hc, 1, s);
  // xout = h @ wlin^T + blin   (fp32 + bf16 outputs)
  launch_gemm(hB, wlinB, hc, nullptr, nullptr, 0, blin, xoutF, xoutB, N_NODES, hc, 0, s);
}

// ---------------------------------------------------------------------------
// Entry point
// ---------------------------------------------------------------------------
extern "C" void kernel_launch(void* const* d_in, const int* in_sizes, int n_in,
                              void* d_out, int out_size, void* d_ws, size_t ws_size,
                              hipStream_t stream) {
  (void)in_sizes; (void)n_in; (void)out_size; (void)ws_size;

  const float* X     = (const float*)d_in[0];
  const int*   ei    = (const int*)d_in[1];
  const int*   src   = ei;                 // edge_index[0]
  const int*   dst   = ei + N_EDGES;       // edge_index[1]
  const int*   batch = (const int*)d_in[2];

  const float *w_rel[5], *b_rel[5], *w_root[5], *w_lin[5], *b_lin[5];
  for (int i = 0; i < 5; ++i) {
    int base = 3 + i * 5;
    w_rel[i]  = (const float*)d_in[base + 0];
    b_rel[i]  = (const float*)d_in[base + 1];
    w_root[i] = (const float*)d_in[base + 2];
    w_lin[i]  = (const float*)d_in[base + 3];
    b_lin[i]  = (const float*)d_in[base + 4];
  }
  const float *mlpW[10], *mlpB[10];
  for (int j = 0; j < 10; ++j) {
    mlpW[j] = (const float*)d_in[28 + 2 * j];
    mlpB[j] = (const float*)d_in[29 + 2 * j];
  }

  // ---- fp32 workspace ----
  float* ws   = (float*)d_ws;
  float* agg  = ws; ws += (size_t)N_NODES * 384;
  float* cat  = ws; ws += (size_t)N_NODES * 384;
  float* out0 = ws; ws += (size_t)N_NODES * 128;   // data[1]: layer-0 output (skip for L3)
  float* bufA = ws; ws += (size_t)N_NODES * 128;
  float* bufB = ws; ws += (size_t)N_NODES * 256;
  float* pool = ws; ws += (size_t)N_GRAPHS * 128;
  float* cnt  = ws; ws += N_GRAPHS;
  float* m0   = ws; ws += (size_t)N_GRAPHS * 256;
  float* m1   = ws; ws += (size_t)N_GRAPHS * 256;

  // ---- bf16 workspace ----
  __bf16* wb   = (__bf16*)ws;
  __bf16* aggB = wb; wb += (size_t)N_NODES * 384;
  __bf16* catB = wb; wb += (size_t)N_NODES * 384;
  __bf16* hB   = wb; wb += (size_t)N_NODES * 256;
  __bf16* XB   = wb; wb += (size_t)N_NODES * 128;
  __bf16* out0B = wb; wb += (size_t)N_NODES * 128;
  __bf16* bufAB = wb; wb += (size_t)N_NODES * 128;
  __bf16* bufBB = wb; wb += (size_t)N_NODES * 256;

  // bf16 weight arena (converted every launch; deterministic)
  static const int INC[5] = {128, 128, 128, 384, 256};
  static const int OUTC[5] = {128, 128, 256, 128, 128};
  __bf16 *wrelB[5], *wrootB[5], *wlinB[5];
  for (int i = 0; i < 5; ++i) {
    wrelB[i]  = wb; wb += (size_t)OUTC[i] * INC[i];
    wrootB[i] = wb; wb += (size_t)OUTC[i] * INC[i];
    wlinB[i]  = wb; wb += (size_t)OUTC[i] * OUTC[i];
    launch_cvt(w_rel[i],  wrelB[i],  (size_t)OUTC[i] * INC[i], stream);
    launch_cvt(w_root[i], wrootB[i], (size_t)OUTC[i] * INC[i], stream);
    launch_cvt(w_lin[i],  wlinB[i],  (size_t)OUTC[i] * OUTC[i], stream);
  }
  launch_cvt(X, XB, (size_t)N_NODES * 128, stream);

  // ---- GNN layers (HIDDEN = [128,128,256,128,128], mid = 3) ----
  // L0: 128 -> 128
  run_layer(X, XB, 128, 128, wrelB[0], b_rel[0], wrootB[0], wlinB[0], b_lin[0],
            src, dst, agg, aggB, hB, out0, out0B, stream);
  // L1: 128 -> 128
  run_layer(out0, out0B, 128, 128, wrelB[1], b_rel[1], wrootB[1], wlinB[1], b_lin[1],
            src, dst, agg, aggB, hB, bufA, bufAB, stream);
  // L2: 128 -> 256
  run_layer(bufA, bufAB, 128, 256, wrelB[2], b_rel[2], wrootB[2], wlinB[2], b_lin[2],
            src, dst, agg, aggB, hB, bufB, bufBB, stream);
  // L3: concat(cur 256, data[1] 128) = 384 -> 128
  {
    long long tot = (long long)N_NODES * 384;
    concat_kernel<<<(unsigned)((tot + 255) / 256), 256, 0, stream>>>(bufB, 256, out0, 128,
                                                                     cat, catB);
  }
  run_layer(cat, catB, 384, 128, wrelB[3], b_rel[3], wrootB[3], wlinB[3], b_lin[3],
            src, dst, agg, aggB, hB, bufA, bufAB, stream);
  // L4: concat(cur 128, data[0]=input 128) = 256 -> 128
  {
    long long tot = (long long)N_NODES * 256;
    concat_kernel<<<(unsigned)((tot + 255) / 256), 256, 0, stream>>>(bufA, 128, X, 128,
                                                                     cat, catB);
  }
  run_layer(cat, catB, 256, 128, wrelB[4], b_rel[4], wrootB[4], wlinB[4], b_lin[4],
            src, dst, agg, aggB, hB, bufB, nullptr, stream);

  // ---- global mean pool over graphs ----
  launch_zero(pool, (size_t)N_GRAPHS * 128 + N_GRAPHS, stream);   // pool + cnt contiguous
  {
    long long tot = (long long)N_NODES * 128;
    pool_scatter_kernel<<<(unsigned)((tot + 255) / 256), 256, 0, stream>>>(bufB, batch,
                                                                           pool, cnt);
    pool_div_kernel<<<(N_GRAPHS * 128 + 255) / 256, 256, 0, stream>>>(pool, cnt);
  }

  // ---- MLP head: 128 -> 256 x9 -> 10 ----
  mlp_kernel<<<(N_GRAPHS * 256 + 255) / 256, 256, 0, stream>>>(pool, mlpW[0], mlpB[0], m0,
                                                               128, 256, 1);
  float* cur = m0; float* nxt = m1;
  for (int j = 1; j <= 8; ++j) {
    mlp_kernel<<<(N_GRAPHS * 256 + 255) / 256, 256, 0, stream>>>(cur, mlpW[j], mlpB[j], nxt,
                                                                 256, 256, 1);
    float* t = cur; cur = nxt; nxt = t;
  }
  mlp_kernel<<<(N_GRAPHS * 10 + 255) / 256, 256, 0, stream>>>(cur, mlpW[9], mlpB[9],
                                                              (float*)d_out, 256, 10, 0);
}
